// RNN_80229989089826
// MI455X (gfx1250) — compile-verified, matrix operand order checked by reading, not run on hardware
//
#include <hip/hip_runtime.h>
#include <math.h>

// ---------------------------------------------------------------------------
// Problem sizes (compile-time constants from the reference)
// ---------------------------------------------------------------------------
#define BT   16384   // B*T = 64*256  (GEMM M dimension)
#define EDIM 128     // E
#define HDIM 512     // H
#define VDIM 8192    // V

typedef __bf16 bf16_t;
typedef __attribute__((ext_vector_type(16))) __bf16 v16bf;
typedef __attribute__((ext_vector_type(8)))  __bf16 v8bf;
typedef __attribute__((ext_vector_type(4)))  __bf16 v4bf;
typedef __attribute__((ext_vector_type(8)))  float  v8f;
typedef __attribute__((ext_vector_type(4)))  float  f4;

// LDS tile geometry for GEMM2: 128x64 bf16 per matrix per buffer,
// row padded 64 -> 72 halves (144B) to spread LDS banks.
#define LDS_PAD_H 72
#define LDS_TILE  (128 * LDS_PAD_H)

// ---------------------------------------------------------------------------
// WMMA 16-bit fragment layout (ISA 7.12.2, wave32):
// lane L: l16 = L&15 (row for A / col for B), hi = L>>4.
// Lane holds K-runs [k + hi*8, +8) and [k + 16 + hi*8, +8).
// ---------------------------------------------------------------------------
__device__ __forceinline__ v16bf frag_bf16(const bf16_t* p) {
    v8bf a = *(const v8bf*)(p);
    v8bf b = *(const v8bf*)(p + 16);
    return __builtin_shufflevector(a, b, 0,1,2,3,4,5,6,7,8,9,10,11,12,13,14,15);
}

__device__ __forceinline__ v16bf frag_f32(const float* __restrict__ p) {
    v8f a = *(const v8f*)(p);
    v8f b = *(const v8f*)(p + 16);
    v16bf r;
#pragma unroll
    for (int e = 0; e < 8; ++e) {
        r[e]     = (__bf16)a[e];
        r[e + 8] = (__bf16)b[e];
    }
    return r;
}

// ---------------------------------------------------------------------------
// CDNA5 async global -> LDS copy (16B per lane), tracked by ASYNCcnt.
// SADDR form: vdst = LDS byte address, vaddr = 32-bit byte offset, saddr = base.
// Flat shared-space addresses: low 32 bits == LDS byte offset (ISA 10.2).
// ---------------------------------------------------------------------------
__device__ __forceinline__ unsigned lds_byte_addr(const void* p) {
    return (unsigned)(uintptr_t)p;
}

__device__ __forceinline__ void async_copy_b128(unsigned lds_addr, const void* gbase,
                                                unsigned byte_off) {
    asm volatile("global_load_async_to_lds_b128 %0, %1, %2"
                 :: "v"(lds_addr), "v"(byte_off), "s"(gbase)
                 : "memory");
}

__device__ __forceinline__ void wait_asynccnt0() {
    asm volatile("s_wait_asynccnt 0" ::: "memory");
}

// ---------------------------------------------------------------------------
// Elementwise fp32 -> bf16 (RNE) for Wy
// ---------------------------------------------------------------------------
__global__ void __launch_bounds__(256)
cvt_f32_to_bf16(const float* __restrict__ in, bf16_t* __restrict__ out, int n) {
    int i = (blockIdx.x * 256 + threadIdx.x) * 4;
    if (i + 3 < n) {
        f4 v = *(const f4*)(in + i);
        v4bf o;
#pragma unroll
        for (int e = 0; e < 4; ++e) o[e] = (__bf16)v[e];
        *(v4bf*)(out + i) = o;
    }
}

// ---------------------------------------------------------------------------
// GEMM1: h[M,H] = tanh( x[M,E] * Wx[H,E]^T + bx + bh ), h stored bf16 (ws).
// 64x64 tile per wave, direct global loads with in-register fp32->bf16.
// K loop NOT unrolled: keeps VGPRs < 256 (round-1 showed s_set_vgpr_msb).
// ---------------------------------------------------------------------------
__global__ void __launch_bounds__(128)
gemm1_tanh(const float* __restrict__ x,  const float* __restrict__ Wx,
           const float* __restrict__ bx, const float* __restrict__ bh,
           bf16_t* __restrict__ h) {
    const int lane = threadIdx.x & 31;
    const int wave = threadIdx.x >> 5;
    const int l16  = lane & 15;
    const int hi   = lane >> 4;

    const int NT   = HDIM / 64;                 // 8 column tiles
    const int tile = blockIdx.x * 4 + wave;     // 2048 tiles total
    if (tile >= (BT / 64) * NT) return;         // wave-uniform guard
    const int mBase = (tile / NT) * 64;
    const int nBase = (tile % NT) * 64;

    v8f acc[4][4] = {};

#pragma unroll 1
    for (int k = 0; k < EDIM; k += 32) {
        v16bf a[4], b[4];
#pragma unroll
        for (int i = 0; i < 4; ++i) {
            a[i] = frag_f32(x  + (size_t)(mBase + i * 16 + l16) * EDIM + k + hi * 8);
            b[i] = frag_f32(Wx + (size_t)(nBase + i * 16 + l16) * EDIM + k + hi * 8);
        }
#pragma unroll
        for (int i = 0; i < 4; ++i)
#pragma unroll
            for (int j = 0; j < 4; ++j)
                acc[i][j] = __builtin_amdgcn_wmma_f32_16x16x32_bf16(
                    false, a[i], false, b[j], (short)0, acc[i][j], false, false);
    }

#pragma unroll
    for (int j = 0; j < 4; ++j) {
        const int n = nBase + j * 16 + l16;
        const float bias = bx[n] + bh[n];
#pragma unroll
        for (int i = 0; i < 4; ++i)
#pragma unroll
            for (int r = 0; r < 8; ++r) {
                const int m = mBase + i * 16 + hi * 8 + r;   // C/D layout: VGPR r -> row
                h[(size_t)m * HDIM + n] = (__bf16)tanhf(acc[i][j][r] + bias);
            }
    }
}

// ---------------------------------------------------------------------------
// GEMM2: out[M,V] = h[M,H](bf16) * Wy[V,H](bf16)^T + by  (fp32 out, NT stores)
//
// Block = 128 threads = 4 waves arranged 2x2; block tile 128x128, wave 64x64.
// K pipelined in 64-wide steps, double-buffered LDS, staged with
// global_load_async_to_lds_b128 (ASYNCcnt) + s_wait_asynccnt + barrier.
// Fragments then come from LDS as ds_load_b128. h+Wy stay hot in L2;
// logits stream out with nontemporal stores so they don't evict them.
// ---------------------------------------------------------------------------
__device__ __forceinline__ void stage_async(bf16_t* sdst, const bf16_t* gsrc,
                                            int rowBase, int k0, int tid) {
    const unsigned lbase = lds_byte_addr(sdst);
#pragma unroll
    for (int c = 0; c < 8; ++c) {
        const int q   = tid + 128 * c;   // 0..1023 chunks of 16B
        const int row = q >> 3;          // 0..127
        const int seg = q & 7;           // 0..7 (8 halves each)
        const unsigned goff  = (unsigned)(((rowBase + row) * HDIM + k0 + seg * 8) * 2);
        const unsigned laddr = lbase + (unsigned)(row * (LDS_PAD_H * 2) + seg * 16);
        async_copy_b128(laddr, gsrc, goff);
    }
}

__global__ void __launch_bounds__(128)
gemm2_bias(const bf16_t* __restrict__ h, const bf16_t* __restrict__ Wy,
           const float* __restrict__ by, float* __restrict__ out) {
    __shared__ bf16_t sA[2][LDS_TILE];   // 2 x 18KB
    __shared__ bf16_t sB[2][LDS_TILE];   // 2 x 18KB

    const int tid  = threadIdx.x;
    const int lane = tid & 31;
    const int wave = tid >> 5;
    const int l16  = lane & 15;
    const int hi   = lane >> 4;
    const int wm   = (wave >> 1) * 64;   // wave row offset in block tile
    const int wn   = (wave & 1) * 64;    // wave col offset in block tile

    const int NBT   = VDIM / 128;                 // 64 column blocks
    const int mBase = (blockIdx.x / NBT) * 128;
    const int nBase = (blockIdx.x % NBT) * 128;

    v8f acc[4][4] = {};

    // Prologue: stage K-step 0 into buffer 0
    stage_async(sA[0], h,  mBase, 0, tid);
    stage_async(sB[0], Wy, nBase, 0, tid);
    wait_asynccnt0();
    __syncthreads();

    const int KSTEPS = HDIM / 64;        // 8
#pragma unroll 1
    for (int s = 0; s < KSTEPS; ++s) {
        const int buf = s & 1;
        if (s + 1 < KSTEPS) {            // stage next step while computing
            stage_async(sA[buf ^ 1], h,  mBase, (s + 1) * 64, tid);
            stage_async(sB[buf ^ 1], Wy, nBase, (s + 1) * 64, tid);
        }

#pragma unroll
        for (int kk = 0; kk < 64; kk += 32) {
            v16bf a[4], b[4];
#pragma unroll
            for (int i = 0; i < 4; ++i) {
                a[i] = frag_bf16(&sA[buf][(wm + i * 16 + l16) * LDS_PAD_H + kk + hi * 8]);
                b[i] = frag_bf16(&sB[buf][(wn + i * 16 + l16) * LDS_PAD_H + kk + hi * 8]);
            }
#pragma unroll
            for (int i = 0; i < 4; ++i)
#pragma unroll
                for (int j = 0; j < 4; ++j)
                    acc[i][j] = __builtin_amdgcn_wmma_f32_16x16x32_bf16(
                        false, a[i], false, b[j], (short)0, acc[i][j], false, false);
        }

        wait_asynccnt0();                // next buffer fully in LDS
        __syncthreads();                 // everyone done reading current buffer
    }

    // Epilogue: bias + nontemporal fp32 stores (don't pollute L2)
#pragma unroll
    for (int j = 0; j < 4; ++j) {
        const int n = nBase + wn + j * 16 + l16;
        const float bias = by[n];
#pragma unroll
        for (int i = 0; i < 4; ++i)
#pragma unroll
            for (int r = 0; r < 8; ++r) {
                const int m = mBase + wm + i * 16 + hi * 8 + r;
                __builtin_nontemporal_store(acc[i][j][r] + bias,
                                            out + (size_t)m * VDIM + n);
            }
    }
}

// ---------------------------------------------------------------------------
// Launch: cvt(Wy) -> gemm1 -> gemm2, all stream-ordered (graph-capture safe).
// Workspace: [0, 8MB)  Wy in bf16 ;  [8MB, 24MB)  h in bf16.
// ---------------------------------------------------------------------------
extern "C" void kernel_launch(void* const* d_in, const int* in_sizes, int n_in,
                              void* d_out, int out_size, void* d_ws, size_t ws_size,
                              hipStream_t stream) {
    const float* x   = (const float*)d_in[0];   // [B,T,E]
    const float* Wx  = (const float*)d_in[1];   // [H,E]
    const float* bx  = (const float*)d_in[2];   // [H]
    // d_in[3] = Wh_w: multiplies zeros in the reference -> unused
    const float* bh  = (const float*)d_in[4];   // [H]
    const float* Wy  = (const float*)d_in[5];   // [V,H]
    const float* by  = (const float*)d_in[6];   // [V]
    float* out = (float*)d_out;                 // [B,T,V] fp32

    bf16_t* WyB = (bf16_t*)d_ws;                                        // 8 MB
    bf16_t* hB  = (bf16_t*)((char*)d_ws + (size_t)VDIM * HDIM * 2);     // 16 MB

    const int nWy = VDIM * HDIM;                // 4,194,304
    cvt_f32_to_bf16<<<nWy / 4 / 256, 256, 0, stream>>>(Wy, WyB, nWy);

    const int g1_tiles = (BT / 64) * (HDIM / 64);   // 2048
    gemm1_tanh<<<g1_tiles / 4, 128, 0, stream>>>(x, Wx, bx, bh, hB);

    const int g2_blocks = (BT / 128) * (VDIM / 128);   // 8192
    gemm2_bias<<<g2_blocks, 128, 0, stream>>>(hB, WyB, by, out);
}
